// HeteroConv_19490561589641
// MI455X (gfx1250) — matrix operand order repaired and checked by reading, not compiled, not run on hardware
//
#include <hip/hip_runtime.h>

#define NN 100000
#define NE 500000
#define HD 128

typedef __attribute__((ext_vector_type(16))) __bf16 bf16x16;
typedef __attribute__((ext_vector_type(8)))  float  f32x8;

static __device__ __forceinline__ __bf16 f2bf(float f) {
    unsigned u = __builtin_bit_cast(unsigned, f);
    unsigned r = (u + 0x7FFFu + ((u >> 16) & 1u)) >> 16;
    return __builtin_bit_cast(__bf16, (unsigned short)r);
}

// ---------------------------------------------------------------------------
// Weight convert + transpose: src f32 [T][K][128] -> dst bf16 [T][128][K]
// ---------------------------------------------------------------------------
__global__ void conv_w(const float* __restrict__ src, __bf16* __restrict__ dst,
                       int T, int K) {
    int idx = blockIdx.x * 256 + threadIdx.x;
    int total = T * K * 128;
    if (idx >= total) return;
    int t = idx / (K * 128);
    int rem = idx - t * K * 128;
    int k = rem >> 7;
    int n = rem & 127;
    dst[((size_t)t * 128 + n) * K + k] = f2bf(src[idx]);
}

// ---------------------------------------------------------------------------
// Segment-sum: msgs[end[e]] += e_feat[e]   (f32 atomics, L2-resident target)
// ---------------------------------------------------------------------------
__global__ void scatter_add(const float* __restrict__ efeat,
                            const int* __restrict__ endI,
                            float* __restrict__ msgs) {
    int idx = blockIdx.x * 256 + threadIdx.x;     // NE*32 threads, 4 floats each
    int eid = idx >> 5;
    if (eid >= NE) return;
    int c = (idx & 31) * 4;
    int dst = endI[eid];
    const float4 v = *(const float4*)(efeat + (size_t)eid * HD + c);
    float* m = msgs + (size_t)dst * HD + c;
    atomicAdd(m + 0, v.x);
    atomicAdd(m + 1, v.y);
    atomicAdd(m + 2, v.z);
    atomicAdd(m + 3, v.w);
}

// ---------------------------------------------------------------------------
// Bucket edges by combo = vid[start] + vid[end]  (0,1,2)
// ---------------------------------------------------------------------------
__global__ void bucket_edges(const int* __restrict__ startI,
                             const int* __restrict__ endI,
                             const int* __restrict__ vid,
                             int* __restrict__ cnt,
                             int* __restrict__ buckets) {
    int eid = blockIdx.x * 256 + threadIdx.x;
    if (eid >= NE) return;
    int c = vid[startI[eid]] + vid[endI[eid]];
    int p = atomicAdd(&cnt[c], 1);
    buckets[(size_t)c * NE + p] = eid;
}

// ---------------------------------------------------------------------------
// WMMA fragment helpers (layouts per CDNA5 ISA §7.12.2)
// ---------------------------------------------------------------------------
// A frag: row M = lane&15; lane-half h: v<4 -> K = 2v + 8h ; v>=4 -> K = 16 + 2(v-4) + 8h
// B frag: col N = lane&15; v<4 -> K = 2v + 8h ; v>=4 -> K = 16 + 2(v-4) + 8h
// D frag: col N = lane&15; VGPR r -> row M = r + 8h

template <int STRIDE>
static __device__ __forceinline__ bf16x16 load_a(const __bf16 (*A)[STRIDE],
                                                 int m, int k0, int half) {
    bf16x16 a;
#pragma unroll
    for (int v = 0; v < 8; ++v) {
        int ka = k0 + ((v < 4) ? (2 * v) : (16 + 2 * (v - 4))) + 8 * half;
        a[2 * v]     = A[m][ka];
        a[2 * v + 1] = A[m][ka + 1];
    }
    return a;
}

static __device__ __forceinline__ bf16x16 load_b(const __bf16* __restrict__ Wnk,
                                                 int nCol, int K, int k0, int half) {
    bf16x16 b;
#pragma unroll
    for (int v = 0; v < 8; ++v) {
        int kb = k0 + ((v < 4) ? (2 * v) : (16 + 2 * (v - 4))) + 8 * half;
        b[2 * v]     = Wnk[(size_t)nCol * K + kb];
        b[2 * v + 1] = Wnk[(size_t)nCol * K + kb + 1];
    }
    return b;
}

// ---------------------------------------------------------------------------
// Node MLP: 16 nodes/block, 256 threads (8 waves), wave w owns cols [16w,16w+16)
// ---------------------------------------------------------------------------
__global__ void __launch_bounds__(256)
node_mlp(const float* __restrict__ x, const float* __restrict__ msgs,
         const int* __restrict__ vid,
         const __bf16* __restrict__ nW1, const float* __restrict__ nb1,
         const float* __restrict__ ng1,  const float* __restrict__ nbt1,
         const __bf16* __restrict__ nW2, const float* __restrict__ nb2,
         const float* __restrict__ ng2,  const float* __restrict__ nbt2,
         float* __restrict__ x_out) {
    __shared__ __bf16 A1[16][264];   // node_in bf16, K=256 (+pad)
    __shared__ float  HF[16][132];
    __shared__ __bf16 A2[16][136];
    __shared__ float  MU[16], RS[16];
    __shared__ int    VIDS[16];

    const int tid  = threadIdx.x;
    const int lane = tid & 31;
    const int w    = tid >> 5;
    const int half = lane >> 4;
    const int l16  = lane & 15;
    const int nCol = w * 16 + l16;
    const int base = blockIdx.x * 16;

    // stage node_in = [x | msgs] as bf16
#pragma unroll
    for (int i = 0; i < 16; ++i) {
        int idx = tid + i * 256;           // 4096 = 16*256
        int r = idx >> 8;
        int c = idx & 255;
        int row = base + r; if (row >= NN) row = NN - 1;
        float v = (c < 128) ? x[(size_t)row * HD + c]
                            : msgs[(size_t)row * HD + (c - 128)];
        A1[r][c] = f2bf(v);
    }
    if (tid < 16) {
        int row = base + tid; if (row >= NN) row = NN - 1;
        VIDS[tid] = vid[row];
    }
    __syncthreads();

    for (int t = 0; t < 2; ++t) {
        // ---- GEMM1: [16x256] x [256x128] ----
        f32x8 acc = {};
        const __bf16* W1t = nW1 + (size_t)t * 128 * 256;
        for (int k0 = 0; k0 < 256; k0 += 32) {
            bf16x16 a = load_a<264>(A1, l16, k0, half);
            bf16x16 b = load_b(W1t, nCol, 256, k0, half);
            acc = __builtin_amdgcn_wmma_f32_16x16x32_bf16(
                      false, a, false, b, (short)0, acc, false, false);
        }
        __syncthreads();
        {
            float bias = nb1[t * HD + nCol];
#pragma unroll
            for (int r = 0; r < 8; ++r) HF[r + 8 * half][nCol] = acc[r] + bias;
        }
        __syncthreads();
        if (tid < 16) {
            float s = 0.f;
            for (int c = 0; c < 128; ++c) s += HF[tid][c];
            float mu = s * (1.f / 128.f);
            float vv = 0.f;
            for (int c = 0; c < 128; ++c) { float d = HF[tid][c] - mu; vv += d * d; }
            MU[tid] = mu; RS[tid] = rsqrtf(vv * (1.f / 128.f) + 1e-5f);
        }
        __syncthreads();
#pragma unroll
        for (int i = 0; i < 8; ++i) {
            int idx = tid + i * 256;       // 2048 = 8*256
            int r = idx >> 7;
            int c = idx & 127;
            float v = (HF[r][c] - MU[r]) * RS[r] * ng1[t * HD + c] + nbt1[t * HD + c];
            A2[r][c] = f2bf(v > 0.f ? v : 0.f);
        }
        __syncthreads();

        // ---- GEMM2: [16x128] x [128x128] ----
        f32x8 acc2 = {};
        const __bf16* W2t = nW2 + (size_t)t * 128 * 128;
        for (int k0 = 0; k0 < 128; k0 += 32) {
            bf16x16 a = load_a<136>(A2, l16, k0, half);
            bf16x16 b = load_b(W2t, nCol, 128, k0, half);
            acc2 = __builtin_amdgcn_wmma_f32_16x16x32_bf16(
                       false, a, false, b, (short)0, acc2, false, false);
        }
        __syncthreads();
        {
            float bias = nb2[t * HD + nCol];
#pragma unroll
            for (int r = 0; r < 8; ++r) HF[r + 8 * half][nCol] = acc2[r] + bias;
        }
        __syncthreads();
        if (tid < 16) {
            float s = 0.f;
            for (int c = 0; c < 128; ++c) s += HF[tid][c];
            float mu = s * (1.f / 128.f);
            float vv = 0.f;
            for (int c = 0; c < 128; ++c) { float d = HF[tid][c] - mu; vv += d * d; }
            MU[tid] = mu; RS[tid] = rsqrtf(vv * (1.f / 128.f) + 1e-5f);
        }
        __syncthreads();
#pragma unroll
        for (int i = 0; i < 8; ++i) {
            int idx = tid + i * 256;
            int r = idx >> 7;
            int c = idx & 127;
            int row = base + r;
            if (row < NN && VIDS[r] == t) {
                float v = tanhf((HF[r][c] - MU[r]) * RS[r] * ng2[t * HD + c] +
                                nbt2[t * HD + c]);
                x_out[(size_t)row * HD + c] = v + x[(size_t)row * HD + c];
            }
        }
        __syncthreads();
    }
}

// ---------------------------------------------------------------------------
// Edge MLP: gridDim.y = combo; 16 bucketed edges/block; K1=384, K2=128
// ---------------------------------------------------------------------------
__global__ void __launch_bounds__(256)
edge_mlp(const float* __restrict__ xout, const float* __restrict__ efeat,
         const int* __restrict__ startI, const int* __restrict__ endI,
         const int* __restrict__ buckets, const int* __restrict__ cnt,
         const __bf16* __restrict__ eW1, const float* __restrict__ eb1,
         const float* __restrict__ eg1,  const float* __restrict__ ebt1,
         const __bf16* __restrict__ eW2, const float* __restrict__ eb2,
         const float* __restrict__ eg2,  const float* __restrict__ ebt2,
         float* __restrict__ e_out) {
    const int combo = blockIdx.y;
    const int cc    = cnt[combo];
    const int base  = blockIdx.x * 16;
    if (base >= cc) return;

    __shared__ __bf16 A1[16][392];   // K=384 (+pad)
    __shared__ float  HF[16][132];
    __shared__ __bf16 A2[16][136];
    __shared__ float  MU[16], RS[16];
    __shared__ int    EID[16], SRC[16], DST[16];

    const int tid  = threadIdx.x;
    const int lane = tid & 31;
    const int w    = tid >> 5;
    const int half = lane >> 4;
    const int l16  = lane & 15;
    const int nCol = w * 16 + l16;

    if (tid < 16) {
        int p = base + tid; if (p >= cc) p = base;
        int eid = buckets[(size_t)combo * NE + p];
        EID[tid] = eid; SRC[tid] = startI[eid]; DST[tid] = endI[eid];
    }
    __syncthreads();

    // stage edge_in = [x_out[src] | x_out[dst] | e] as bf16 : 16x384 = 6144
#pragma unroll
    for (int i = 0; i < 24; ++i) {
        int idx = tid + i * 256;
        int r = idx / 384;
        int c = idx - r * 384;
        float v;
        if (c < 128)       v = xout[(size_t)SRC[r] * HD + c];
        else if (c < 256)  v = xout[(size_t)DST[r] * HD + (c - 128)];
        else               v = efeat[(size_t)EID[r] * HD + (c - 256)];
        A1[r][c] = f2bf(v);
    }
    __syncthreads();

    // ---- GEMM1: [16x384] x [384x128] ----
    f32x8 acc = {};
    const __bf16* W1c = eW1 + (size_t)combo * 128 * 384;
    for (int k0 = 0; k0 < 384; k0 += 32) {
        bf16x16 a = load_a<392>(A1, l16, k0, half);
        bf16x16 b = load_b(W1c, nCol, 384, k0, half);
        acc = __builtin_amdgcn_wmma_f32_16x16x32_bf16(
                  false, a, false, b, (short)0, acc, false, false);
    }
    {
        float bias = eb1[combo * HD + nCol];
#pragma unroll
        for (int r = 0; r < 8; ++r) HF[r + 8 * half][nCol] = acc[r] + bias;
    }
    __syncthreads();
    if (tid < 16) {
        float s = 0.f;
        for (int c = 0; c < 128; ++c) s += HF[tid][c];
        float mu = s * (1.f / 128.f);
        float vv = 0.f;
        for (int c = 0; c < 128; ++c) { float d = HF[tid][c] - mu; vv += d * d; }
        MU[tid] = mu; RS[tid] = rsqrtf(vv * (1.f / 128.f) + 1e-5f);
    }
    __syncthreads();
#pragma unroll
    for (int i = 0; i < 8; ++i) {
        int idx = tid + i * 256;
        int r = idx >> 7;
        int c = idx & 127;
        float v = (HF[r][c] - MU[r]) * RS[r] * eg1[combo * HD + c] + ebt1[combo * HD + c];
        A2[r][c] = f2bf(v > 0.f ? v : 0.f);
    }
    __syncthreads();

    // ---- GEMM2: [16x128] x [128x128] ----
    f32x8 acc2 = {};
    const __bf16* W2c = eW2 + (size_t)combo * 128 * 128;
    for (int k0 = 0; k0 < 128; k0 += 32) {
        bf16x16 a = load_a<136>(A2, l16, k0, half);
        bf16x16 b = load_b(W2c, nCol, 128, k0, half);
        acc2 = __builtin_amdgcn_wmma_f32_16x16x32_bf16(
                   false, a, false, b, (short)0, acc2, false, false);
    }
    __syncthreads();
    {
        float bias = eb2[combo * HD + nCol];
#pragma unroll
        for (int r = 0; r < 8; ++r) HF[r + 8 * half][nCol] = acc2[r] + bias;
    }
    __syncthreads();
    if (tid < 16) {
        float s = 0.f;
        for (int c = 0; c < 128; ++c) s += HF[tid][c];
        float mu = s * (1.f / 128.f);
        float vv = 0.f;
        for (int c = 0; c < 128; ++c) { float d = HF[tid][c] - mu; vv += d * d; }
        MU[tid] = mu; RS[tid] = rsqrtf(vv * (1.f / 128.f) + 1e-5f);
    }
    __syncthreads();
#pragma unroll
    for (int i = 0; i < 8; ++i) {
        int idx = tid + i * 256;
        int r = idx >> 7;
        int c = idx & 127;
        if (base + r < cc) {
            int eid = EID[r];
            float v = tanhf((HF[r][c] - MU[r]) * RS[r] * eg2[combo * HD + c] +
                            ebt2[combo * HD + c]);
            e_out[(size_t)eid * HD + c] = v + efeat[(size_t)eid * HD + c];
        }
    }
}

// ---------------------------------------------------------------------------
extern "C" void kernel_launch(void* const* d_in, const int* in_sizes, int n_in,
                              void* d_out, int out_size, void* d_ws, size_t ws_size,
                              hipStream_t stream) {
    (void)in_sizes; (void)n_in; (void)out_size; (void)ws_size;

    const float* x       = (const float*)d_in[0];
    const float* e       = (const float*)d_in[1];
    const int*   eidx    = (const int*)d_in[2];
    const int*   startI  = eidx;
    const int*   endI    = eidx + NE;
    const int*   vid     = (const int*)d_in[3];
    const float* node_W1 = (const float*)d_in[4];
    const float* node_b1 = (const float*)d_in[5];
    const float* node_g1 = (const float*)d_in[6];
    const float* node_bt1= (const float*)d_in[7];
    const float* node_W2 = (const float*)d_in[8];
    const float* node_b2 = (const float*)d_in[9];
    const float* node_g2 = (const float*)d_in[10];
    const float* node_bt2= (const float*)d_in[11];
    const float* edge_W1 = (const float*)d_in[12];
    const float* edge_b1 = (const float*)d_in[13];
    const float* edge_g1 = (const float*)d_in[14];
    const float* edge_bt1= (const float*)d_in[15];
    const float* edge_W2 = (const float*)d_in[16];
    const float* edge_b2 = (const float*)d_in[17];
    const float* edge_g2 = (const float*)d_in[18];
    const float* edge_bt2= (const float*)d_in[19];

    float* x_out = (float*)d_out;
    float* e_out = (float*)d_out + (size_t)NN * HD;

    char* ws = (char*)d_ws;
    float* msgs = (float*)ws;
    size_t off = (size_t)NN * HD * sizeof(float);
    int* buckets = (int*)(ws + off);  off += (size_t)3 * NE * sizeof(int);
    int* cnt     = (int*)(ws + off);  off += 64;
    __bf16* nW1  = (__bf16*)(ws + off); off += (size_t)2 * 256 * 128 * 2;
    __bf16* nW2  = (__bf16*)(ws + off); off += (size_t)2 * 128 * 128 * 2;
    __bf16* eW1  = (__bf16*)(ws + off); off += (size_t)3 * 384 * 128 * 2;
    __bf16* eW2  = (__bf16*)(ws + off); off += (size_t)3 * 128 * 128 * 2;

    hipMemsetAsync(msgs, 0, (size_t)NN * HD * sizeof(float), stream);
    hipMemsetAsync(cnt, 0, 3 * sizeof(int), stream);

    conv_w<<<(2 * 256 * 128 + 255) / 256, 256, 0, stream>>>(node_W1, nW1, 2, 256);
    conv_w<<<(2 * 128 * 128 + 255) / 256, 256, 0, stream>>>(node_W2, nW2, 2, 128);
    conv_w<<<(3 * 384 * 128 + 255) / 256, 256, 0, stream>>>(edge_W1, eW1, 3, 384);
    conv_w<<<(3 * 128 * 128 + 255) / 256, 256, 0, stream>>>(edge_W2, eW2, 3, 128);

    scatter_add<<<(NE * 32 + 255) / 256, 256, 0, stream>>>(e, endI, msgs);
    bucket_edges<<<(NE + 255) / 256, 256, 0, stream>>>(startI, endI, vid, cnt, buckets);

    node_mlp<<<(NN + 15) / 16, 256, 0, stream>>>(
        x, msgs, vid, nW1, node_b1, node_g1, node_bt1,
        nW2, node_b2, node_g2, node_bt2, x_out);

    edge_mlp<<<dim3((NE + 15) / 16, 3), 256, 0, stream>>>(
        x_out, e, startI, endI, buckets, cnt,
        eW1, edge_b1, edge_g1, edge_bt1,
        eW2, edge_b2, edge_g2, edge_bt2, e_out);
}
